// VectorQuantizerEMA_84009560310081
// MI455X (gfx1250) — matrix operand order repaired and checked by reading, not compiled, not run on hardware
//
#include <hip/hip_runtime.h>

// ---------------------------------------------------------------------------
// VQ-VAE EMA vector quantizer for gfx1250 (MI455X).
// Heavy op: [65536,128] x [128,4096] distance matmul -> argmin with
// v_wmma_f32_16x16x32_bf16 (bf16 hi/lo split: 3 WMMAs per 32-K slice ~ f32
// accuracy at bf16 matrix-core rate). 32-code codebook tiles are double-
// buffered in LDS via global_load_async_to_lds_b128 (ASYNCcnt, issued by
// waves 0/1) and shared by all 8 waves: 24 WMMAs per barrier interval.
// ---------------------------------------------------------------------------

#define VQ_D 128
#define VQ_DECAY 0.9f
#define VQ_EPS 1e-5f

// LDS tile layout: 32 rows x 128 ushorts, padded to 136 ushorts/row so the 16
// lanes of a half-wave (same column, rows r..r+15) hit disjoint 4-bank groups.
#define TILE_CODES 32
#define ROW_STRIDE 136                        // ushorts (272 B, 16B-aligned)
#define HALF_BLOCK (TILE_CODES * ROW_STRIDE)  // 4352 ushorts (hi or lo block)
#define BUF_USHORTS (2 * HALF_BLOCK)          // 8704 ushorts per buffer

typedef __attribute__((ext_vector_type(16))) __bf16 v16bf;
typedef __attribute__((ext_vector_type(8)))  __bf16 v8bf;
typedef __attribute__((ext_vector_type(8)))  float  v8f;

union Frag16 { v16bf v; v8bf h[2]; };

struct alignas(8) US4 { unsigned short x, y, z, w; };

__device__ __forceinline__ unsigned short bf16_rne(float f) {
  unsigned u = __float_as_uint(f);
  u += 0x7FFFu + ((u >> 16) & 1u);
  return (unsigned short)(u >> 16);
}
__device__ __forceinline__ float bf16_up(unsigned short h) {
  return __uint_as_float(((unsigned)h) << 16);
}

// CDNA5 async copy: global -> LDS, 16 bytes per lane, tracked by ASYNCcnt.
__device__ __forceinline__ void async_load_b128(unsigned lds_byte_addr,
                                                const void* gaddr) {
  asm volatile("global_load_async_to_lds_b128 %0, %1, off"
               :: "v"(lds_byte_addr), "v"(gaddr)
               : "memory");
}
__device__ __forceinline__ void wait_asynccnt0() {
  asm volatile("s_wait_asynccnt 0" ::: "memory");
}

// ---------------------------------------------------------------------------
// 0) zero scratch accumulators + the diff output cell
// ---------------------------------------------------------------------------
__global__ void vq_zero(float* __restrict__ dw, float* __restrict__ bcs,
                        float* __restrict__ diff, int KD, int K) {
  int i = blockIdx.x * blockDim.x + threadIdx.x;
  if (i < KD) dw[i] = 0.0f;
  if (i < K)  bcs[i] = 0.0f;
  if (i == 0) *diff = 0.0f;
}

// ---------------------------------------------------------------------------
// 1) z -> bf16 hi/lo split (vectorized float4 -> 2x ushort4)
// ---------------------------------------------------------------------------
__global__ void vq_conv_z(const float* __restrict__ z,
                          unsigned short* __restrict__ zh,
                          unsigned short* __restrict__ zl, int nd4) {
  int i = blockIdx.x * blockDim.x + threadIdx.x;
  if (i >= nd4) return;
  float4 v = reinterpret_cast<const float4*>(z)[i];
  US4 h, l;
  h.x = bf16_rne(v.x); l.x = bf16_rne(v.x - bf16_up(h.x));
  h.y = bf16_rne(v.y); l.y = bf16_rne(v.y - bf16_up(h.y));
  h.z = bf16_rne(v.z); l.z = bf16_rne(v.z - bf16_up(h.z));
  h.w = bf16_rne(v.w); l.w = bf16_rne(v.w - bf16_up(h.w));
  reinterpret_cast<US4*>(zh)[i] = h;
  reinterpret_cast<US4*>(zl)[i] = l;
}

// ---------------------------------------------------------------------------
// 2) embedding -> bf16 hi/lo + exact f32 row norms ||e||^2 (one wave per row)
// ---------------------------------------------------------------------------
__global__ void __launch_bounds__(256) vq_conv_e(const float* __restrict__ e,
                                                 unsigned short* __restrict__ eh,
                                                 unsigned short* __restrict__ el,
                                                 float* __restrict__ en2, int K) {
  int lane = threadIdx.x & 31;
  int wave = threadIdx.x >> 5;
  int k = blockIdx.x * 8 + wave;
  if (k >= K) return;
  float4 v = reinterpret_cast<const float4*>(e + (size_t)k * VQ_D)[lane];
  US4 h, l;
  h.x = bf16_rne(v.x); l.x = bf16_rne(v.x - bf16_up(h.x));
  h.y = bf16_rne(v.y); l.y = bf16_rne(v.y - bf16_up(h.y));
  h.z = bf16_rne(v.z); l.z = bf16_rne(v.z - bf16_up(h.z));
  h.w = bf16_rne(v.w); l.w = bf16_rne(v.w - bf16_up(h.w));
  reinterpret_cast<US4*>(eh + (size_t)k * VQ_D)[lane] = h;
  reinterpret_cast<US4*>(el + (size_t)k * VQ_D)[lane] = l;
  float s = v.x * v.x + v.y * v.y + v.z * v.z + v.w * v.w;
  #pragma unroll
  for (int off = 16; off >= 1; off >>= 1) s += __shfl_xor(s, off, 32);
  if (lane == 0) en2[k] = s;
}

// ---------------------------------------------------------------------------
// 3) WMMA distance + argmin.
//    Block = 256 threads = 8 waves; each wave owns 16 rows; grid covers N/128.
//    dist(m, code) = ||e||^2 - 2 * (z . e)    (||z||^2 dropped: row-constant)
//    32-code tiles stream through LDS with async-to-LDS double buffering.
// ---------------------------------------------------------------------------
__global__ void __launch_bounds__(256) vq_argmin(
    const unsigned short* __restrict__ zh, const unsigned short* __restrict__ zl,
    const unsigned short* __restrict__ eh, const unsigned short* __restrict__ el,
    const float* __restrict__ en2, int* __restrict__ idx_i,
    float* __restrict__ idx_f, int K) {
  __shared__ __align__(32) unsigned short lds[2 * BUF_USHORTS];

  const int lane = threadIdx.x & 31;
  const int wave = threadIdx.x >> 5;
  const int rowBase = blockIdx.x * 128 + wave * 16;
  const int m   = lane & 15;           // A row / B column within a 16-col tile
  const int aKb = (lane < 16) ? 0 : 16;// A-fragment K base (contiguous k = 16h + s order)
  const int bKb = (lane < 16) ? 0 : 16;// B-fragment K base

  const unsigned ldsBase = (unsigned)(size_t)(void*)lds;   // wave-relative LDS addr

  // A fragments for this wave's 16 rows: 4 k-chunks x {hi, lo}
  Frag16 ah[4], al[4];
  const unsigned short* zrh = zh + (size_t)(rowBase + m) * VQ_D;
  const unsigned short* zrl = zl + (size_t)(rowBase + m) * VQ_D;
  #pragma unroll
  for (int kc = 0; kc < 4; ++kc) {
    ah[kc].h[0] = *reinterpret_cast<const v8bf*>(zrh + kc * 32 + aKb);
    ah[kc].h[1] = *reinterpret_cast<const v8bf*>(zrh + kc * 32 + aKb + 8);
    al[kc].h[0] = *reinterpret_cast<const v8bf*>(zrl + kc * 32 + aKb);
    al[kc].h[1] = *reinterpret_cast<const v8bf*>(zrl + kc * 32 + aKb + 8);
  }

  float rmin[8];
  int   ridx[8];
  #pragma unroll
  for (int v = 0; v < 8; ++v) { rmin[v] = 3.4e38f; ridx[v] = 0; }

  const int nTiles = K / TILE_CODES;

  // --- async stage of one 32-code tile: wave 0 ships the hi block, wave 1
  //     the lo block (512 x 16B chunks each -> 16 instrs/lane).
  //     chunk c in [0,512): row = c>>4, col = (c&15)*8 ushorts (padded dst).
  auto issue_tile = [&](int tile, int buf) {
    const unsigned short* gsrc =
        (wave == 0 ? eh : el) + (size_t)tile * TILE_CODES * VQ_D;
    const unsigned bufBase = ldsBase + (unsigned)buf * BUF_USHORTS * 2u +
                             (wave == 0 ? 0u : HALF_BLOCK * 2u);
    #pragma unroll
    for (int j = 0; j < 16; ++j) {
      const int c = j * 32 + lane;
      const int row = c >> 4;
      const int col = (c & 15) * 8;                // ushorts
      const unsigned dst = (unsigned)(row * ROW_STRIDE + col) * 2u;  // bytes
      async_load_b128(bufBase + dst, gsrc + c * 8);
    }
  };

  if (wave < 2) {             // wave-uniform branch: EXEC stays all-ones
    issue_tile(0, 0);
    wait_asynccnt0();
  }
  __syncthreads();

  for (int tile = 0; tile < nTiles; ++tile) {
    const int cur = tile & 1;
    if (tile + 1 < nTiles && wave < 2) issue_tile(tile + 1, cur ^ 1);

    const unsigned short* lbBase = lds + (size_t)cur * BUF_USHORTS;
    v8f acc[2];

    #pragma unroll
    for (int t = 0; t < 2; ++t) {      // two 16-column WMMA tiles per LDS tile
      const unsigned short* lb =
          lbBase + (size_t)(t * 16 + m) * ROW_STRIDE + bKb;

      Frag16 bh[4], bl[4];
      #pragma unroll
      for (int kc = 0; kc < 4; ++kc) {
        bh[kc].h[0] = *reinterpret_cast<const v8bf*>(lb + kc * 32);
        bh[kc].h[1] = *reinterpret_cast<const v8bf*>(lb + kc * 32 + 8);
        bl[kc].h[0] = *reinterpret_cast<const v8bf*>(lb + HALF_BLOCK + kc * 32);
        bl[kc].h[1] = *reinterpret_cast<const v8bf*>(lb + HALF_BLOCK + kc * 32 + 8);
      }

      v8f c = {};
      #pragma unroll
      for (int kc = 0; kc < 4; ++kc) {
        c = __builtin_amdgcn_wmma_f32_16x16x32_bf16(false, ah[kc].v, false, bh[kc].v,
                                                    (short)0, c, false, false);
        c = __builtin_amdgcn_wmma_f32_16x16x32_bf16(false, al[kc].v, false, bh[kc].v,
                                                    (short)0, c, false, false);
        c = __builtin_amdgcn_wmma_f32_16x16x32_bf16(false, ah[kc].v, false, bl[kc].v,
                                                    (short)0, c, false, false);
      }
      acc[t] = c;
    }

    #pragma unroll
    for (int t = 0; t < 2; ++t) {
      const int code = tile * TILE_CODES + t * 16 + m;
      const float e2 = en2[code];
      #pragma unroll
      for (int v = 0; v < 8; ++v) {
        float d = e2 - 2.0f * acc[t][v];
        if (d < rmin[v] || (d == rmin[v] && code < ridx[v])) {
          rmin[v] = d; ridx[v] = code;
        }
      }
    }

    if (wave < 2) wait_asynccnt0();   // next buffer resident before anyone reads
    __syncthreads();
  }

  // Cross-lane argmin within each 16-lane half (xor masks stay inside halves)
  #pragma unroll
  for (int v = 0; v < 8; ++v) {
    float d = rmin[v]; int ix = ridx[v];
    #pragma unroll
    for (int off = 8; off >= 1; off >>= 1) {
      float od = __shfl_xor(d, off, 32);
      int   oi = __shfl_xor(ix, off, 32);
      if (od < d || (od == d && oi < ix)) { d = od; ix = oi; }
    }
    rmin[v] = d; ridx[v] = ix;
  }
  if (m == 0) {
    const int hi = lane >> 4;
    #pragma unroll
    for (int v = 0; v < 8; ++v) {
      int row = rowBase + v + 8 * hi;  // C/D layout: M = vgpr + 8*(lane>=16)
      idx_i[row] = ridx[v];
      idx_f[row] = (float)ridx[v];
    }
  }
}

// ---------------------------------------------------------------------------
// 4) segment sums: batch_cluster_size and dw, via f32 global atomics
// ---------------------------------------------------------------------------
__global__ void vq_scatter(const float* __restrict__ z, const int* __restrict__ idx,
                           float* __restrict__ dw, float* __restrict__ bcs, int ND) {
  int i = blockIdx.x * blockDim.x + threadIdx.x;
  if (i >= ND) return;
  int r = i >> 7;
  int d = i & (VQ_D - 1);
  int k = idx[r];
  atomicAdd(&dw[(size_t)k * VQ_D + d], z[i]);
  if (d == 0) atomicAdd(&bcs[k], 1.0f);
}

// ---------------------------------------------------------------------------
// 5) EMA updates
// ---------------------------------------------------------------------------
__global__ void vq_ema(const float* __restrict__ ema_w, const float* __restrict__ cluster,
                       const float* __restrict__ dw, const float* __restrict__ bcs,
                       float* __restrict__ o_emaw, float* __restrict__ o_cs,
                       int K, int KD) {
  int i = blockIdx.x * blockDim.x + threadIdx.x;
  if (i < KD) o_emaw[i] = ema_w[i] * VQ_DECAY + (1.0f - VQ_DECAY) * dw[i];
  if (i < K)  o_cs[i]   = cluster[i] * VQ_DECAY + (1.0f - VQ_DECAY) * bcs[i];
}

// ---------------------------------------------------------------------------
// 6) n = sum(cluster_size_new)  (single block)
// ---------------------------------------------------------------------------
__global__ void __launch_bounds__(256) vq_nsum(const float* __restrict__ cs,
                                               float* __restrict__ scal, int K) {
  __shared__ float red[256];
  float s = 0.0f;
  for (int i = threadIdx.x; i < K; i += 256) s += cs[i];
  red[threadIdx.x] = s;
  __syncthreads();
  for (int st = 128; st > 0; st >>= 1) {
    if (threadIdx.x < st) red[threadIdx.x] += red[threadIdx.x + st];
    __syncthreads();
  }
  if (threadIdx.x == 0) scal[0] = red[0];
}

// ---------------------------------------------------------------------------
// 7) Laplace-smoothed normalize
// ---------------------------------------------------------------------------
__global__ void vq_norm(const float* __restrict__ o_emaw, const float* __restrict__ o_cs,
                        const float* __restrict__ scal, float* __restrict__ o_embn,
                        int K, int KD) {
  int i = blockIdx.x * blockDim.x + threadIdx.x;
  if (i >= KD) return;
  int k = i >> 7;
  float n = scal[0];
  float cs = (o_cs[k] + VQ_EPS) / (n + (float)K * VQ_EPS) * n;
  o_embn[i] = o_emaw[i] / cs;
}

// ---------------------------------------------------------------------------
// 8) gather z_q (straight-through value == z_q) + diff = 2*mean((z_q-z)^2)
// ---------------------------------------------------------------------------
__global__ void __launch_bounds__(256) vq_gather(const float* __restrict__ z,
                                                 const int* __restrict__ idx,
                                                 const float* __restrict__ embn,
                                                 float* __restrict__ o_zq,
                                                 float* __restrict__ o_diff,
                                                 int ND, float scale) {
  int i = blockIdx.x * blockDim.x + threadIdx.x;
  float part = 0.0f;
  if (i < ND) {
    int r = i >> 7;
    int d = i & (VQ_D - 1);
    float q = embn[(size_t)idx[r] * VQ_D + d];
    o_zq[i] = q;
    float df = q - z[i];
    part = df * df;
  }
  __shared__ float red[256];
  red[threadIdx.x] = part;
  __syncthreads();
  for (int st = 128; st > 0; st >>= 1) {
    if (threadIdx.x < st) red[threadIdx.x] += red[threadIdx.x + st];
    __syncthreads();
  }
  if (threadIdx.x == 0) atomicAdd(o_diff, red[0] * scale);
}

// ---------------------------------------------------------------------------
extern "C" void kernel_launch(void* const* d_in, const int* in_sizes, int n_in,
                              void* d_out, int out_size, void* d_ws, size_t ws_size,
                              hipStream_t stream) {
  const int D = VQ_D;
  const int N = in_sizes[0] / D;   // 65536
  const int K = in_sizes[1] / D;   // 4096
  const int ND = N * D;
  const int KD = K * D;

  const float* z       = (const float*)d_in[0];
  const float* emb     = (const float*)d_in[1];
  const float* cluster = (const float*)d_in[2];
  const float* ema_w   = (const float*)d_in[3];

  // outputs, concatenated in reference return order
  float* o_zq   = (float*)d_out;          // N*D
  float* o_idx  = o_zq + (size_t)ND;      // N
  float* o_diff = o_idx + (size_t)N;      // 1
  float* o_cs   = o_diff + 1;             // K
  float* o_emaw = o_cs + (size_t)K;       // K*D
  float* o_embn = o_emaw + (size_t)KD;    // K*D

  // workspace carve-up (all sizes multiples of 256B)
  char* w = (char*)d_ws;
  unsigned short* zh = (unsigned short*)w; w += (size_t)ND * 2;
  unsigned short* zl = (unsigned short*)w; w += (size_t)ND * 2;
  unsigned short* eh = (unsigned short*)w; w += (size_t)KD * 2;
  unsigned short* el = (unsigned short*)w; w += (size_t)KD * 2;
  float* dw   = (float*)w; w += (size_t)KD * 4;
  float* en2  = (float*)w; w += (size_t)K * 4;
  float* bcs  = (float*)w; w += (size_t)K * 4;
  int*   idxs = (int*)w;   w += (size_t)N * 4;
  float* scal = (float*)w; w += 256;

  const int T = 256;
  vq_zero   <<<(KD + T - 1) / T, T, 0, stream>>>(dw, bcs, o_diff, KD, K);
  vq_conv_z <<<(ND / 4 + T - 1) / T, T, 0, stream>>>(z, zh, zl, ND / 4);
  vq_conv_e <<<(K + 7) / 8, T, 0, stream>>>(emb, eh, el, en2, K);
  vq_argmin <<<N / 128, T, 0, stream>>>(zh, zl, eh, el, en2, idxs, o_idx, K);
  vq_scatter<<<(ND + T - 1) / T, T, 0, stream>>>(z, idxs, dw, bcs, ND);
  vq_ema    <<<(KD + T - 1) / T, T, 0, stream>>>(ema_w, cluster, dw, bcs, o_emaw, o_cs, K, KD);
  vq_nsum   <<<1, T, 0, stream>>>(o_cs, scal, K);
  vq_norm   <<<(KD + T - 1) / T, T, 0, stream>>>(o_emaw, o_cs, scal, o_embn, K, KD);
  vq_gather <<<(ND + T - 1) / T, T, 0, stream>>>(z, idxs, o_embn, o_zq, o_diff, ND,
                                                 2.0f / (float)ND);
}
